// SDPAttention2D_28647431864474
// MI455X (gfx1250) — matrix operand order repaired and checked by reading, not compiled, not run on hardware
//
#include <hip/hip_runtime.h>

typedef __attribute__((ext_vector_type(16))) __bf16 v16bf;
typedef __attribute__((ext_vector_type(8)))  float  v8f;
typedef int v4i __attribute__((vector_size(16)));   // matches builtin param type

__device__ __forceinline__ unsigned short f2bf(float f) {
    union { float f; unsigned int u; } x; x.f = f;
    unsigned int u = x.u;
    return (unsigned short)((u + 0x7FFFu + ((u >> 16) & 1u)) >> 16);
}

// 16-byte global -> LDS copy: async (ASYNCcnt) when available, sync fallback.
#if __has_builtin(__builtin_amdgcn_global_load_async_to_lds_b128)
#define HAVE_ASYNC_LDS 1
__device__ __forceinline__ void g2l16(void* dst, const void* src) {
    __builtin_amdgcn_global_load_async_to_lds_b128((v4i*)src, (v4i*)dst, 0, 0);
}
__device__ __forceinline__ void g2l_wait() {
#if __has_builtin(__builtin_amdgcn_s_wait_asynccnt)
    __builtin_amdgcn_s_wait_asynccnt(0);
#else
    asm volatile("s_wait_asynccnt 0" ::: "memory");
#endif
}
#else
#define HAVE_ASYNC_LDS 0
__device__ __forceinline__ void g2l16(void* dst, const void* src) {
    *(uint4*)dst = *(const uint4*)src;
}
__device__ __forceinline__ void g2l_wait() {}
#endif

// ---------------------------------------------------------------------------
// fp32 -> bf16 bulk converter (runs once per tensor; removes cvt from GEMM loop)
// ---------------------------------------------------------------------------
__global__ __launch_bounds__(256)
void cvt_bf16(const float* __restrict__ in, unsigned short* __restrict__ out, int n)
{
    int i = (blockIdx.x * 256 + threadIdx.x) * 4;
    if (i < n) {
        float4 f = *(const float4*)(in + i);
        unsigned int lo = (unsigned int)f2bf(f.x) | ((unsigned int)f2bf(f.y) << 16);
        unsigned int hi = (unsigned int)f2bf(f.z) | ((unsigned int)f2bf(f.w) << 16);
        *(uint2*)(out + i) = make_uint2(lo, hi);
    }
}

// ---------------------------------------------------------------------------
// C[M,N] = A[M,K] * B[N,K]^T, bf16 inputs, fp32 out.
// Block 64x64, 8 waves of 16x32, K step 32, async LDS fill + prefetch.
// A tile stored with 8-half chunks permuted {0,2,1,3} so an A fragment is one
// contiguous v16bf; B tile permuted the same way.
// ---------------------------------------------------------------------------
__global__ __launch_bounds__(256)
void gemm_bt_bf16(const unsigned short* __restrict__ A, const unsigned short* __restrict__ Bw,
                  float* __restrict__ C, int M, int N, int K)
{
    __shared__ unsigned short Xt[64][32];   // chunk-swizzled A tile
    __shared__ unsigned short Wt[64][32];   // B tile (chunk-swizzled, same as A)

    const int tid  = threadIdx.x;
    const int lane = tid & 31;
    const int w    = tid >> 5;
    const int wm   = w & 3;
    const int wn   = w >> 2;
    const int m0   = blockIdx.y * 64;
    const int n0   = blockIdx.x * 64;

    const int row = tid >> 2;          // 0..63
    const int ct  = tid & 3;           // chunk of 8 halves
    const int pct = ((ct & 1) << 1) | (ct >> 1);   // {0,2,1,3} permutation

    const size_t arow = (size_t)(m0 + row) * K + ct * 8;
    const size_t brow = (size_t)(n0 + row) * K + ct * 8;

    v8f acc0 = {}, acc1 = {};

    for (int kb = 0; kb < K; kb += 32) {
        g2l16(&Xt[row][pct * 8], A + arow + kb);
        g2l16(&Wt[row][pct * 8], Bw + brow + kb);
        if (kb + 32 < K) {
            __builtin_prefetch(A + arow + kb + 32, 0, 0);
            __builtin_prefetch(Bw + brow + kb + 32, 0, 0);
        }
        g2l_wait();
        __syncthreads();

        const int half16 = (lane & 16) ? 16 : 0;
        v16bf af = *(const v16bf*)&Xt[wm * 16 + (lane & 15)][half16];
        v16bf b0 = *(const v16bf*)&Wt[wn * 32 + (lane & 15)][half16];
        v16bf b1 = *(const v16bf*)&Wt[wn * 32 + 16 + (lane & 15)][half16];

        acc0 = __builtin_amdgcn_wmma_f32_16x16x32_bf16(false, af, false, b0,
                                                       (short)0, acc0, false, false);
        acc1 = __builtin_amdgcn_wmma_f32_16x16x32_bf16(false, af, false, b1,
                                                       (short)0, acc1, false, false);
        __syncthreads();
    }

    const int mo = (lane & 16) ? 8 : 0;
#pragma unroll
    for (int r = 0; r < 8; ++r) {
        int m = m0 + wm * 16 + r + mo;
        int n = n0 + wn * 32 + (lane & 15);
        C[(size_t)m * N + n]      = acc0[r];
        C[(size_t)m * N + n + 16] = acc1[r];
    }
}

// ---------------------------------------------------------------------------
// RMSNorm + RoPE. q written natural to (B,nH,L,64);
// k natural (B,nH,L,64); v transposed (B,nH,64,L). All bf16.
// ---------------------------------------------------------------------------
__global__ __launch_bounds__(256)
void normrope(const float* __restrict__ qkv, const float* __restrict__ qg,
              const float* __restrict__ kg,
              unsigned short* __restrict__ qn, unsigned short* __restrict__ kn,
              unsigned short* __restrict__ vt)
{
    const int lane = threadIdx.x & 31;
    const int gid  = blockIdx.x * 8 + (threadIdx.x >> 5); // (b*1024+l)*12 + h
    const int h = gid % 12;
    const int l = (gid / 12) & 1023;
    const int b = gid / (12 * 1024);
    const size_t base = (size_t)(b * 1024 + l) * 2304 + h * 64;
    const int bh = b * 12 + h;

    float inv_freq = __powf(10000.0f, -(float)lane * (1.0f / 32.0f));
    float sa, ca;
    __sincosf((float)l * inv_freq, &sa, &ca);

    // natural position for q
    const int pq = lane;

#pragma unroll
    for (int s = 0; s < 2; ++s) {           // s=0: q, s=1: k
        const float* g = s ? kg : qg;
        float x1 = qkv[base + s * 768 + lane];
        float x2 = qkv[base + s * 768 + 32 + lane];
        float ss = x1 * x1 + x2 * x2;
#pragma unroll
        for (int m = 1; m < 32; m <<= 1) ss += __shfl_xor(ss, m, 32);
        float rs = rsqrtf(ss * (1.0f / 64.0f) + 1e-6f);
        float y1 = x1 * rs * g[lane], y2 = x2 * rs * g[lane + 32];
        float o1 = y1 * ca - y2 * sa, o2 = y1 * sa + y2 * ca;
        size_t ob = ((size_t)bh * 1024 + l) * 64;
        if (s == 0) {
            qn[ob + pq]      = f2bf(o1);
            qn[ob + 32 + pq] = f2bf(o2);
        } else {
            kn[ob + lane]      = f2bf(o1);
            kn[ob + 32 + lane] = f2bf(o2);
        }
    }
    {   // v: transpose into (bh, hd, L)
        float v1 = qkv[base + 1536 + lane];
        float v2 = qkv[base + 1536 + 32 + lane];
        size_t ob = (size_t)bh * 64 * 1024 + l;
        vt[ob + (size_t)lane * 1024]        = f2bf(v1);
        vt[ob + (size_t)(lane + 32) * 1024] = f2bf(v2);
    }
}

// ---------------------------------------------------------------------------
// Flash attention, bf16 WMMA, online softmax. Output bf16 (B,L,nH,64).
// ---------------------------------------------------------------------------
__global__ __launch_bounds__(256)
void flash_attn(const unsigned short* __restrict__ qn,
                const unsigned short* __restrict__ kn,
                const unsigned short* __restrict__ vt,
                unsigned short* __restrict__ out)
{
    __shared__ unsigned short Kt[32][64];     // [key][hd]
    __shared__ unsigned short Vt[64][32];     // [hd][key]
    __shared__ unsigned short Ps[8][16][32];  // per-wave P staging, [row][key]

    const int tid  = threadIdx.x;
    const int lane = tid & 31;
    const int w    = tid >> 5;
    const int bh   = blockIdx.y;
    const int b = bh / 12, h = bh % 12;
    const int q0 = blockIdx.x * 128 + w * 16;

    // Q fragments: direct contiguous v16bf loads (qn natural layout)
    v16bf aq[2];
    {
        const size_t rb = ((size_t)bh * 1024 + q0 + (lane & 15)) * 64;
        const int half16 = (lane & 16) ? 16 : 0;
        aq[0] = *(const v16bf*)(qn + rb + half16);
        aq[1] = *(const v16bf*)(qn + rb + 32 + half16);
    }

    v8f acc[4] = {{}, {}, {}, {}};
    float rmax[8], rsum[8];
#pragma unroll
    for (int r = 0; r < 8; ++r) { rmax[r] = -1e30f; rsum[r] = 0.0f; }

    // cooperative fill indices (one 16B chunk per thread per tile)
    const int kkey = tid >> 3, kdc = tid & 7;                 // Kt: 32x64
    const int vd = tid >> 2, vkc = tid & 3;                   // Vt: 64x32
    const size_t kbase = ((size_t)bh * 1024 + kkey) * 64 + kdc * 8;
    const size_t vbase = ((size_t)bh * 64 + vd) * 1024 + vkc * 8;

    for (int kb = 0; kb < 1024; kb += 32) {
        g2l16(&Kt[kkey][kdc * 8], kn + kbase + (size_t)kb * 64);
        g2l16(&Vt[vd][vkc * 8],   vt + vbase + kb);
        if (kb + 32 < 1024) {
            __builtin_prefetch(kn + kbase + (size_t)(kb + 32) * 64, 0, 0);
            __builtin_prefetch(vt + vbase + kb + 32, 0, 0);
        }
        g2l_wait();
        __syncthreads();

        // S = Q * K^T
        v8f s0 = {}, s1 = {};
#pragma unroll
        for (int ks = 0; ks < 2; ++ks) {
            const int hb = ks * 32 + ((lane & 16) ? 16 : 0);
            v16bf bk0 = *(const v16bf*)&Kt[(lane & 15)][hb];
            v16bf bk1 = *(const v16bf*)&Kt[16 + (lane & 15)][hb];
            s0 = __builtin_amdgcn_wmma_f32_16x16x32_bf16(false, aq[ks], false, bk0,
                                                         (short)0, s0, false, false);
            s1 = __builtin_amdgcn_wmma_f32_16x16x32_bf16(false, aq[ks], false, bk1,
                                                         (short)0, s1, false, false);
        }

        // online softmax; write P into staging ([row][key])
        const int Mo = (lane & 16) ? 8 : 0;
        const int sp0 = (lane & 15);        // col of N=lane%16
        const int sp1 = 16 + (lane & 15);   // col of N=16+lane%16
        float alpha[8];
#pragma unroll
        for (int r = 0; r < 8; ++r) {
            float a = s0[r] * 0.125f, c = s1[r] * 0.125f;
            float tm = fmaxf(a, c);
#pragma unroll
            for (int m = 1; m < 16; m <<= 1) tm = fmaxf(tm, __shfl_xor(tm, m, 32));
            float nm = fmaxf(rmax[r], tm);
            float al = __expf(rmax[r] - nm);
            float p0 = __expf(a - nm), p1 = __expf(c - nm);
            float ts = p0 + p1;
#pragma unroll
            for (int m = 1; m < 16; m <<= 1) ts += __shfl_xor(ts, m, 32);
            rsum[r] = rsum[r] * al + ts;
            rmax[r] = nm;
            alpha[r] = al;
            Ps[w][r + Mo][sp0] = f2bf(p0);
            Ps[w][r + Mo][sp1] = f2bf(p1);
        }
#pragma unroll
        for (int nf = 0; nf < 4; ++nf)
#pragma unroll
            for (int r = 0; r < 8; ++r) acc[nf][r] *= alpha[r];

        __syncthreads();

        // P fragment: single contiguous v16bf (staging is natural [row][key])
        v16bf ap = *(const v16bf*)&Ps[w][lane & 15][(lane & 16) ? 16 : 0];

        const int ko = (lane & 16) ? 16 : 0;
#pragma unroll
        for (int nf = 0; nf < 4; ++nf) {
            v16bf bv = *(const v16bf*)&Vt[nf * 16 + (lane & 15)][ko];
            acc[nf] = __builtin_amdgcn_wmma_f32_16x16x32_bf16(false, ap, false, bv,
                                                              (short)0, acc[nf], false, false);
        }
        __syncthreads();
    }

    // normalize and store bf16 (B,L,nH*64)
    const int Mo = (lane & 16) ? 8 : 0;
#pragma unroll
    for (int r = 0; r < 8; ++r) {
        const float inv = 1.0f / rsum[r];
        const int q = q0 + r + Mo;
        const size_t ob = (size_t)(b * 1024 + q) * 768 + h * 64;
#pragma unroll
        for (int nf = 0; nf < 4; ++nf)
            out[ob + nf * 16 + (lane & 15)] = f2bf(acc[nf][r] * inv);
    }
}

// ---------------------------------------------------------------------------
extern "C" void kernel_launch(void* const* d_in, const int* in_sizes, int n_in,
                              void* d_out, int out_size, void* d_ws, size_t ws_size,
                              hipStream_t stream)
{
    const float* x     = (const float*)d_in[0];  // (8,32,32,768)
    const float* w_qkv = (const float*)d_in[1];  // (2304,768)
    const float* qg    = (const float*)d_in[2];
    const float* kg    = (const float*)d_in[3];
    const float* w_out = (const float*)d_in[4];  // (768,768)
    float* out = (float*)d_out;

    char* ws = (char*)d_ws;
    unsigned short* xb  = (unsigned short*)ws;                  // 12.6 MB
    unsigned short* wqb = (unsigned short*)(ws + 12582912);     //  3.5 MB
    unsigned short* wob = (unsigned short*)(ws + 16121856);     //  1.2 MB
    float* qkv          = (float*)(ws + 17301504);              // 75.5 MB
    unsigned short* qn  = (unsigned short*)(ws + 92798976);     // 12.6 MB
    unsigned short* kn  = (unsigned short*)(ws + 105381888);    // 12.6 MB
    unsigned short* vt  = (unsigned short*)(ws + 117964800);    // 12.6 MB
    unsigned short* aob = (unsigned short*)(ws + 17301504);     // aliases qkv (dead)

    // one-shot fp32 -> bf16 conversions
    cvt_bf16<<<dim3(6144), 256, 0, stream>>>(x,     xb,  6291456);
    cvt_bf16<<<dim3(1728), 256, 0, stream>>>(w_qkv, wqb, 1769472);
    cvt_bf16<<<dim3(576),  256, 0, stream>>>(w_out, wob, 589824);

    // qkv = x @ w_qkv^T
    gemm_bt_bf16<<<dim3(36, 128), 256, 0, stream>>>(xb, wqb, qkv, 8192, 2304, 768);
    // rmsnorm + rope + repack
    normrope<<<dim3(12288), 256, 0, stream>>>(qkv, qg, kg, qn, kn, vt);
    // flash attention
    flash_attn<<<dim3(8, 96), 256, 0, stream>>>(qn, kn, vt, aob);
    // final projection
    gemm_bt_bf16<<<dim3(12, 128), 256, 0, stream>>>(aob, wob, out, 8192, 768, 768);
}
